// BaseFormer_8297876815905
// MI455X (gfx1250) — matrix operand order repaired and checked by reading, not compiled, not run on hardware
//
#include <hip/hip_runtime.h>
#include <cstdint>
#include <cstddef>

typedef __bf16 bf16;
typedef __attribute__((ext_vector_type(16))) __bf16 v16bf;
typedef __attribute__((ext_vector_type(8)))  __bf16 v8bf;
typedef __attribute__((ext_vector_type(8)))  float  v8f;

#define N_TOK 4096          // per-batch spatial size (64*64) == 4*1024 patch tokens
#define LDSS  40            // LDS k-row stride: 32 K-elems + 8 pad (80B rows, 16B aligned)

__device__ __forceinline__ float silu_f(float v) { return v / (1.0f + __expf(-v)); }

// ---------------------------------------------------------------- prep kernels
__global__ void k_cvt(const float* __restrict__ s, bf16* __restrict__ d, int n) {
    int i = blockIdx.x * 256 + threadIdx.x;
    if (i < n) d[i] = (bf16)s[i];
}

// conv3_w (O,C,3,3) fp32 -> wT[tap][o][c] bf16 (contiguous in c for A-frags)
__global__ void k_conv3T(const float* __restrict__ w, bf16* __restrict__ wT) {
    int i = blockIdx.x * 256 + threadIdx.x;
    if (i >= 9 * 256 * 256) return;
    int tap = i >> 16, o = (i >> 8) & 255, c = i & 255;
    int dy = tap / 3, dx = tap % 3;
    wT[i] = (bf16)w[(((o * 256 + c) * 3) + dy) * 3 + dx];
}

__global__ void k_bnprep(const float* __restrict__ g, const float* __restrict__ b,
                         const float* __restrict__ m, const float* __restrict__ v,
                         float eps, float* __restrict__ sc, float* __restrict__ sh, int C) {
    int i = blockIdx.x * 256 + threadIdx.x;
    if (i < C) { float s = g[i] * rsqrtf(v[i] + eps); sc[i] = s; sh[i] = b[i] - m[i] * s; }
}

// ---------------------------------------------------------------- WMMA fragment helpers
// A (16x32 bf16): lane m=l&15, h=l>>4; elem e -> k = h*8 + (e<8 ? e : e+8)
__device__ __forceinline__ v16bf ld_afrag(const bf16* __restrict__ p) {
    v8bf lo = *(const v8bf*)(p);
    v8bf hi = *(const v8bf*)(p + 16);
    v16bf a;
#pragma unroll
    for (int j = 0; j < 8; ++j) { a[j] = lo[j]; a[j + 8] = hi[j]; }
    return a;
}
// B (32x16 bf16) from swizzled LDS row (rot = row & 24, k stored at (k+rot)&31)
__device__ __forceinline__ v16bf ld_bfrag(const bf16* __restrict__ row, int rot, int hl) {
    v8bf lo = *(const v8bf*)(row + (((hl * 8) + rot) & 31));
    v8bf hi = *(const v8bf*)(row + (((hl * 8) + 16 + rot) & 31));
    v16bf b;
#pragma unroll
    for (int j = 0; j < 8; ++j) { b[j] = lo[j]; b[j + 8] = hi[j]; }
    return b;
}
__device__ __forceinline__ v8f wmma_acc(v16bf a, v16bf b, v8f c) {
    return __builtin_amdgcn_wmma_f32_16x16x32_bf16(false, a, false, b, (short)0, c, false, false);
}

// ---------------------------------------------------------------- generic GEMM
// Out[b] = act(W(MxK) * X[b](K x 4096) [+bias] [*scale+shift]) ; 32 M-rows per block.
// OUTMODE: 0=store bf16 [b][m][n]; 1=store f32 patch-permuted (conv1->p);
//          2=f32 accumulate into Out (residual); 3=store f32 direct.
// ACT: 0=none; 1=silu; 2=relu for m>=reluStart (qkv: v-rows only).
template <int OUTMODE, int ACT>
__global__ __launch_bounds__(256) void gemm_ws(const bf16* __restrict__ W,
                                               const bf16* __restrict__ X,
                                               const float* __restrict__ bias,
                                               const float* __restrict__ scale,
                                               const float* __restrict__ shift,
                                               void* __restrict__ Out,
                                               int M, int K, int reluStart) {
    __shared__ __align__(16) bf16 xs[128 * LDSS];
    const int b = blockIdx.z;
    const int mBase = blockIdx.y * 32;
    const int nBase = blockIdx.x * 128;
    const int t = threadIdx.x;
    const int lane = t & 31, wave = t >> 5;
    const int hl = lane >> 4, l16 = lane & 15;
    const bf16* Xb = X + (size_t)b * K * N_TOK;
    v8f acc0 = {}, acc1 = {};

    for (int k0 = 0; k0 < K; k0 += 32) {
        __syncthreads();
#pragma unroll
        for (int i = 0; i < 2; ++i) {              // stage 32x128 tile, transposed+swizzled
            int id8 = t + 256 * i;                 // 512 16B chunks
            int kk  = id8 >> 4;
            int nn8 = (id8 & 15) << 3;
            const bf16* src = Xb + (size_t)(k0 + kk) * N_TOK + nBase + nn8;
            v8bf chunk = *(const v8bf*)src;
            if (k0 + 32 < K) __builtin_prefetch(src + 32 * N_TOK, 0, 3);
            int q = (kk + (nn8 & 24)) & 31;        // k-block rotation per LDS row group
#pragma unroll
            for (int j = 0; j < 8; ++j) xs[(nn8 + j) * LDSS + q] = chunk[j];
        }
        __syncthreads();
        const int nloc = wave * 16 + l16;
        v16bf bf_ = ld_bfrag(xs + nloc * LDSS, nloc & 24, hl);
        const bf16* w0 = W + (size_t)(mBase + l16) * K + k0 + hl * 8;
        v16bf a0 = ld_afrag(w0);
        v16bf a1 = ld_afrag(w0 + (size_t)16 * K);
        acc0 = wmma_acc(a0, bf_, acc0);
        acc1 = wmma_acc(a1, bf_, acc1);
    }

#pragma unroll
    for (int mt = 0; mt < 2; ++mt) {
        v8f acc = mt ? acc1 : acc0;
#pragma unroll
        for (int r = 0; r < 8; ++r) {
            int m = mBase + mt * 16 + r + 8 * hl;
            int n = nBase + wave * 16 + l16;
            float v = acc[r];
            if (bias)  v += bias[m];
            if (scale) v = v * scale[m] + shift[m];
            if (ACT == 1) v = silu_f(v);
            if (ACT == 2 && m >= reluStart) v = fmaxf(v, 0.0f);
            if (OUTMODE == 0) {
                ((bf16*)Out)[((size_t)b * M + m) * N_TOK + n] = (bf16)v;
            } else if (OUTMODE == 1) {
                int h = n >> 6, w = n & 63;
                int np = ((h & 1) * 2 + (w & 1)) * 1024 + (h >> 1) * 32 + (w >> 1);
                ((float*)Out)[((size_t)b * M + m) * N_TOK + np] = v;
            } else if (OUTMODE == 2) {
                float* p = (float*)Out + ((size_t)b * M + m) * N_TOK + n;
                *p = *p + v;
            } else {
                ((float*)Out)[((size_t)b * M + m) * N_TOK + n] = v;
            }
        }
    }
}

// ---------------------------------------------------------------- 3x3 conv (implicit GEMM) + BN + SiLU
// One LDS halo (4 full image rows x 32 K) per K-slice serves all 9 taps: 18 WMMAs per stage.
// LDS layout: xh[row][k] with row = r*66 + (w+1), r=0..3 (h0-1..h0+2), w pad cols 0 & 65 = 0.
__global__ __launch_bounds__(256) void conv3_gemm(const bf16* __restrict__ wT,   // [9][256][256]
                                                  const bf16* __restrict__ Xb16, // [B][256][4096]
                                                  const float* __restrict__ sc,
                                                  const float* __restrict__ sh,
                                                  bf16* __restrict__ Out) {
    __shared__ __align__(16) bf16 xh[4 * 66 * LDSS];   // 21120 B
    const int b = blockIdx.z;
    const int mBase = blockIdx.y * 32;
    const int nBase = blockIdx.x * 128;
    const int h0 = blockIdx.x * 2;
    const int t = threadIdx.x;
    const int lane = t & 31, wave = t >> 5;
    const int hl = lane >> 4, l16 = lane & 15;
    const bf16* Xb = Xb16 + (size_t)b * 256 * N_TOK;

    { // zero the halo pad columns (wp = 0 and 65) once, all 32 k slots
        int r = t >> 6, c = (t >> 5) & 1, k = t & 31;
        xh[((r * 66) + (c ? 65 : 0)) * LDSS + k] = (bf16)0.0f;
    }
    v8f acc0 = {}, acc1 = {};

    for (int k0 = 0; k0 < 256; k0 += 32) {
        __syncthreads();
#pragma unroll
        for (int i = 0; i < 4; ++i) {              // stage 32k x 4rows x 64w halo
            int id8 = t + 256 * i;                 // 1024 16B chunks
            int k   = id8 >> 5;
            int rw8 = id8 & 31;
            int r   = rw8 >> 3;
            int w8  = (rw8 & 7) << 3;
            int hg  = h0 - 1 + r;
            v8bf chunk = {};
            if ((unsigned)hg < 64u) {
                const bf16* src = Xb + (size_t)(k0 + k) * N_TOK + hg * 64 + w8;
                chunk = *(const v8bf*)src;
                if (k0 + 32 < 256) __builtin_prefetch(src + 32 * N_TOK, 0, 3);
            }
            int rowBase = r * 66 + w8 + 1;         // (rowBase+j)&24 == rowBase&24? no: use per-j row
#pragma unroll
            for (int j = 0; j < 8; ++j) {
                int row = rowBase + j;
                xh[row * LDSS + ((k + (row & 24)) & 31)] = chunk[j];
            }
        }
        __syncthreads();
        const int nloc = wave * 16 + l16;
#pragma unroll
        for (int tap = 0; tap < 9; ++tap) {
            int rh  = (nloc >> 6) + tap / 3;       // halo row (uniform per wave)
            int wp  = (nloc & 63) + tap % 3;       // halo col incl. +1 pad shift
            int row = rh * 66 + wp;
            v16bf bf_ = ld_bfrag(xh + row * LDSS, row & 24, hl);
            const bf16* w0 = wT + (size_t)tap * 65536 + (size_t)(mBase + l16) * 256 + k0 + hl * 8;
            v16bf a0 = ld_afrag(w0);
            v16bf a1 = ld_afrag(w0 + 16 * 256);
            acc0 = wmma_acc(a0, bf_, acc0);
            acc1 = wmma_acc(a1, bf_, acc1);
        }
    }
#pragma unroll
    for (int mt = 0; mt < 2; ++mt) {
        v8f acc = mt ? acc1 : acc0;
#pragma unroll
        for (int r = 0; r < 8; ++r) {
            int m = mBase + mt * 16 + r + 8 * hl;
            int n = nBase + wave * 16 + l16;
            float v = silu_f(acc[r] * sc[m] + sh[m]);
            Out[((size_t)b * 256 + m) * N_TOK + n] = (bf16)v;
        }
    }
}

// ---------------------------------------------------------------- elementwise / reductions
__global__ void k_bnbf16(const float* __restrict__ p, const float* __restrict__ sc,
                         const float* __restrict__ sh, bf16* __restrict__ y, int total) {
    int i = blockIdx.x * 256 + threadIdx.x;
    if (i >= total) return;
    int d = (i >> 12) & 127;
    y[i] = (bf16)(p[i] * sc[d] + sh[d]);
}

// q[b][n] = qkv_w[0,:] . y[b,:,n] + qkv_b[0]
__global__ void k_q(const bf16* __restrict__ y, const float* __restrict__ qkvw,
                    const float* __restrict__ qkvb, float* __restrict__ q) {
    int n = blockIdx.x * 256 + threadIdx.x;
    int b = blockIdx.y;
    const bf16* yp = y + (size_t)b * 128 * N_TOK + n;
    float s = qkvb[0];
#pragma unroll 4
    for (int d = 0; d < 128; ++d) s += (float)yp[(size_t)d * N_TOK] * qkvw[d];
    q[b * N_TOK + n] = s;
}

// in-place softmax over 1024 tokens per (b,t)
__global__ void k_softmax(float* __restrict__ q) {
    __shared__ float red[256];
    float* base = q + blockIdx.x * 1024;
    int t = threadIdx.x;
    float v0[4], mx = -1e30f;
#pragma unroll
    for (int i = 0; i < 4; ++i) { v0[i] = base[t + 256 * i]; mx = fmaxf(mx, v0[i]); }
    red[t] = mx; __syncthreads();
    for (int s = 128; s > 0; s >>= 1) { if (t < s) red[t] = fmaxf(red[t], red[t + s]); __syncthreads(); }
    mx = red[0]; __syncthreads();
    float sum = 0.0f;
#pragma unroll
    for (int i = 0; i < 4; ++i) { v0[i] = __expf(v0[i] - mx); sum += v0[i]; }
    red[t] = sum; __syncthreads();
    for (int s = 128; s > 0; s >>= 1) { if (t < s) red[t] += red[t + s]; __syncthreads(); }
    float inv = 1.0f / red[0];
#pragma unroll
    for (int i = 0; i < 4; ++i) base[t + 256 * i] = v0[i] * inv;
}

// ctx[b,d,t] = sum_l k[b,d,t*1024+l] * scores[b,t,l]  (one wave32 per row)
__global__ void k_ctx(const bf16* __restrict__ kv, const float* __restrict__ scores,
                      float* __restrict__ ctx) {
    int row = blockIdx.x * 8 + (threadIdx.x >> 5);
    int lane = threadIdx.x & 31;
    int b = row >> 9, d = (row >> 2) & 127, tt = row & 3;
    const bf16*  kp = kv + ((size_t)(b * 256 + d)) * N_TOK + tt * 1024;
    const float* sp = scores + (b * 4 + tt) * 1024;
    float s = 0.0f;
    for (int l = lane; l < 1024; l += 32) s += (float)kp[l] * sp[l];
#pragma unroll
    for (int off = 16; off; off >>= 1) s += __shfl_down(s, off, 32);
    if (lane == 0) ctx[(b * 128 + d) * 4 + tt] = s;
}

// a[b,d,n] = relu(v)[b,d,n] * ctx[b,d,n/1024]   (v-rows already relu'd in qkv epilogue)
__global__ void k_act(const bf16* __restrict__ kv, const float* __restrict__ ctx,
                      bf16* __restrict__ a, int total) {
    int i = blockIdx.x * 256 + threadIdx.x;
    if (i >= total) return;
    int n = i & 4095, d = (i >> 12) & 127, b = i >> 19;
    float v = (float)kv[((size_t)(b * 256 + 128 + d)) * N_TOK + n];
    a[i] = (bf16)(v * ctx[(b * 128 + d) * 4 + (n >> 10)]);
}

// final BN + un-patch: y[b,d,h*64+w] = p[b,d,np]*sc+sh
__global__ void k_final(const float* __restrict__ p, const float* __restrict__ sc,
                        const float* __restrict__ sh, bf16* __restrict__ y, int total) {
    int i = blockIdx.x * 256 + threadIdx.x;
    if (i >= total) return;
    int n = i & 4095, d = (i >> 12) & 127, b = i >> 19;
    int h = n >> 6, w = n & 63;
    int np = ((h & 1) * 2 + (w & 1)) * 1024 + (h >> 1) * 32 + (w >> 1);
    y[i] = (bf16)(p[((size_t)(b * 128 + d)) * N_TOK + np] * sc[d] + sh[d]);
}

// ---------------------------------------------------------------- host
extern "C" void kernel_launch(void* const* d_in, const int* in_sizes, int n_in,
                              void* d_out, int out_size, void* d_ws, size_t ws_size,
                              hipStream_t stream) {
    (void)in_sizes; (void)n_in; (void)out_size;
    // input order = setup_inputs() insertion order (dict order, DFS):
    // 0:x 1:conv3_w 2-5:bn3(g,b,m,v) 6:conv1_w
    // blocks[k] @ 7+16k: 0-3 attn_bn(g,b,m,v) 4:qkv_w 5:qkv_b 6:out_w 7:out_b
    //                    8-11 ffn_bn(g,b,m,v) 12:ffn1_w 13:ffn1_b 14:ffn2_w 15:ffn2_b
    // 39-42:final_bn 43:proj_w 44-47:proj_bn
    const float* inF[48];
    for (int i = 0; i < 48; ++i) inF[i] = (const float*)d_in[i];

    const int SZ_X = 16 * 256 * N_TOK;   // 16,777,216
    const int SZ_P = 16 * 128 * N_TOK;   //  8,388,608

    char* base = (char*)d_ws;
    size_t off = 0;
    auto alloc = [&](size_t bytes) -> void* {
        void* p = base + off;
        off = (off + bytes + 255) & ~(size_t)255;
        return p;
    };
    bf16*  A    = (bf16*)alloc((size_t)SZ_X * 2);          // x(bf16) / yb / abuf
    bf16*  Bb   = (bf16*)alloc((size_t)SZ_X * 2);          // h1 / kv / ffn-mid
    float* P    = (float*)alloc((size_t)SZ_P * 4);         // residual stream (patch layout)
    float* qbuf = (float*)alloc((size_t)16 * N_TOK * 4);   // q -> softmax in place
    float* ctxb = (float*)alloc((size_t)16 * 128 * 4 * 4);
    bf16*  c3T  = (bf16*)alloc((size_t)9 * 256 * 256 * 2);
    bf16*  w1   = (bf16*)alloc((size_t)128 * 256 * 2);
    bf16*  wqkv[2], *wout[2], *wf1[2], *wf2[2];
    for (int k = 0; k < 2; ++k) {
        wqkv[k] = (bf16*)alloc((size_t)257 * 128 * 2);
        wout[k] = (bf16*)alloc((size_t)128 * 128 * 2);
        wf1[k]  = (bf16*)alloc((size_t)256 * 128 * 2);
        wf2[k]  = (bf16*)alloc((size_t)128 * 256 * 2);
    }
    bf16*  wproj = (bf16*)alloc((size_t)256 * 128 * 2);
    float* bn3s = (float*)alloc(256 * 4); float* bn3t = (float*)alloc(256 * 4);
    float* abns[2], *abnt[2], *fbns[2], *fbnt[2];
    for (int k = 0; k < 2; ++k) {
        abns[k] = (float*)alloc(128 * 4); abnt[k] = (float*)alloc(128 * 4);
        fbns[k] = (float*)alloc(128 * 4); fbnt[k] = (float*)alloc(128 * 4);
    }
    float* fins = (float*)alloc(128 * 4); float* fint = (float*)alloc(128 * 4);
    float* pjs  = (float*)alloc(256 * 4); float* pjt  = (float*)alloc(256 * 4);
    if (off > ws_size) return;   // workspace too small; deterministic no-op

    dim3 blk(256);
    auto g1 = [](int n) { return dim3((n + 255) / 256); };

    // ---- parameter prep
    k_cvt<<<g1(SZ_X), blk, 0, stream>>>(inF[0], A, SZ_X);
    k_conv3T<<<g1(9 * 256 * 256), blk, 0, stream>>>(inF[1], c3T);
    k_cvt<<<g1(128 * 256), blk, 0, stream>>>(inF[6], w1, 128 * 256);
    for (int k = 0; k < 2; ++k) {
        int p0 = 7 + 16 * k;
        k_cvt<<<g1(257 * 128), blk, 0, stream>>>(inF[p0 + 4],  wqkv[k], 257 * 128);
        k_cvt<<<g1(128 * 128), blk, 0, stream>>>(inF[p0 + 6],  wout[k], 128 * 128);
        k_cvt<<<g1(256 * 128), blk, 0, stream>>>(inF[p0 + 12], wf1[k],  256 * 128);
        k_cvt<<<g1(128 * 256), blk, 0, stream>>>(inF[p0 + 14], wf2[k],  128 * 256);
        k_bnprep<<<1, blk, 0, stream>>>(inF[p0 + 0], inF[p0 + 1], inF[p0 + 2], inF[p0 + 3],
                                        1e-5f, abns[k], abnt[k], 128);
        k_bnprep<<<1, blk, 0, stream>>>(inF[p0 + 8], inF[p0 + 9], inF[p0 + 10], inF[p0 + 11],
                                        1e-5f, fbns[k], fbnt[k], 128);
    }
    k_cvt<<<g1(256 * 128), blk, 0, stream>>>(inF[43], wproj, 256 * 128);
    k_bnprep<<<1, blk, 0, stream>>>(inF[2],  inF[3],  inF[4],  inF[5],  1e-3f, bn3s, bn3t, 256);
    k_bnprep<<<1, blk, 0, stream>>>(inF[39], inF[40], inF[41], inF[42], 1e-5f, fins, fint, 128);
    k_bnprep<<<1, blk, 0, stream>>>(inF[44], inF[45], inF[46], inF[47], 1e-3f, pjs, pjt, 256);

    // ---- conv3 (implicit GEMM, all taps per stage) + bn3 + silu : A -> Bb
    conv3_gemm<<<dim3(32, 8, 16), blk, 0, stream>>>(c3T, A, bn3s, bn3t, Bb);
    // ---- conv1 -> residual stream P (fp32, patch layout)
    gemm_ws<1, 0><<<dim3(32, 4, 16), blk, 0, stream>>>(w1, Bb, nullptr, nullptr, nullptr,
                                                       P, 128, 256, 1 << 30);
    // ---- two attention/FFN blocks
    for (int k = 0; k < 2; ++k) {
        int p0 = 7 + 16 * k;
        k_bnbf16<<<g1(SZ_P), blk, 0, stream>>>(P, abns[k], abnt[k], A, SZ_P);
        // k,v rows (qkv rows 1..256), v-rows (m>=128) relu'd in epilogue
        gemm_ws<0, 2><<<dim3(32, 8, 16), blk, 0, stream>>>(wqkv[k] + 128, A, inF[p0 + 5] + 1,
                                                           nullptr, nullptr, Bb, 256, 128, 128);
        k_q<<<dim3(16, 16), blk, 0, stream>>>(A, inF[p0 + 4], inF[p0 + 5], qbuf);
        k_softmax<<<dim3(64), blk, 0, stream>>>(qbuf);
        k_ctx<<<dim3(1024), blk, 0, stream>>>(Bb, qbuf, ctxb);
        k_act<<<g1(SZ_P), blk, 0, stream>>>(Bb, ctxb, A, SZ_P);
        // out projection, p += y
        gemm_ws<2, 0><<<dim3(32, 4, 16), blk, 0, stream>>>(wout[k], A, inF[p0 + 7],
                                                           nullptr, nullptr, P, 128, 128, 0);
        // FFN
        k_bnbf16<<<g1(SZ_P), blk, 0, stream>>>(P, fbns[k], fbnt[k], A, SZ_P);
        gemm_ws<0, 1><<<dim3(32, 8, 16), blk, 0, stream>>>(wf1[k], A, inF[p0 + 13],
                                                           nullptr, nullptr, Bb, 256, 128, 0);
        gemm_ws<2, 0><<<dim3(32, 4, 16), blk, 0, stream>>>(wf2[k], Bb, inF[p0 + 15],
                                                           nullptr, nullptr, P, 128, 256, 0);
    }
    // ---- final_bn + un-patch -> A, then proj + proj_bn -> d_out (fp32)
    k_final<<<g1(SZ_P), blk, 0, stream>>>(P, fins, fint, A, SZ_P);
    gemm_ws<3, 0><<<dim3(32, 8, 16), blk, 0, stream>>>(wproj, A, nullptr, pjs, pjt,
                                                       (float*)d_out, 256, 128, 0);
}
